// Style_dubber_model_15_SPost_Duration_21655225106571
// MI455X (gfx1250) — matrix operand order repaired and checked by reading, not compiled, not run on hardware
//
#include <hip/hip_runtime.h>

#define DEV __device__ __forceinline__

typedef __bf16 bf16_t;
typedef __attribute__((ext_vector_type(8)))  __bf16 v8bf;
typedef __attribute__((ext_vector_type(16))) __bf16 v16bf;
typedef __attribute__((ext_vector_type(8)))  float  v8f;
typedef __attribute__((ext_vector_type(2)))  unsigned u32x2;

static constexpr int BB = 512;   // batch
static constexpr int KK = 20;    // refs per channel

// ---------------- scalar helpers ----------------
DEV unsigned bfbits(float f) {               // f32 -> bf16 bits (RNE)
  unsigned u = __builtin_bit_cast(unsigned, f);
  return (u + 0x7FFFu + ((u >> 16) & 1u)) >> 16;
}
DEV unsigned pack2(float a, float b) {       // two bf16 packed in a dword
  return (bfbits(a) & 0xFFFFu) | (bfbits(b) << 16);
}
DEV bf16_t f2bf(float f) {
  unsigned short h = (unsigned short)bfbits(f);
  return __builtin_bit_cast(bf16_t, h);
}

// ---------------- WMMA fragment loaders (bf16, 16x16x32) ----------------
// A (16x32): lane L holds row mo+(L&15); K = ko+half*8+{0..7} and ko+16+half*8+{0..7}
DEV v16bf load_a_frag(const bf16_t* base, int pitch, int mo, int ko, int lane) {
  int half = lane >> 4;
  int m = mo + (lane & 15);
  const bf16_t* p = base + m * pitch + ko + half * 8;
  v8bf lo = *(const v8bf*)(p);
  v8bf hi = *(const v8bf*)(p + 16);
  return __builtin_shufflevector(lo, hi, 0,1,2,3,4,5,6,7,8,9,10,11,12,13,14,15);
}
// B (32x16) from LDS: lane L holds col N=no+(L&15); K = ko+half*8+{0..7} and ko+16+half*8+{0..7}
DEV v16bf load_b_frag(const bf16_t* base, int pitch, int ko, int no, int lane) {
  int half = lane >> 4;
  const bf16_t* p = base + (ko + half * 8) * pitch + no + (lane & 15);
  v16bf r;
  for (int s = 0; s < 8; s++) { r[s] = p[s * pitch]; r[s + 8] = p[(s + 16) * pitch]; }
  return r;
}
// B (32x16) straight from global bf16 weights: base points at row ko, col no
DEV v16bf load_b_frag_g(const bf16_t* base, int lane) {
  int half = lane >> 4;
  const bf16_t* p = base + (size_t)(half * 8) * 256 + (lane & 15);
  v16bf r;
  for (int s = 0; s < 8; s++) { r[s] = p[s * 256]; r[s + 8] = p[(s + 16) * 256]; }
  return r;
}
DEV v8f wmma_bf16(v16bf a, v16bf b, v8f c) {
  return __builtin_amdgcn_wmma_f32_16x16x32_bf16(false, a, false, b, (short)0, c, false, false);
}

// relation label of edge (i,j) in the fixed RGAT graph (0 = no edge)
DEV int adj_label(int i, int j, int stage) {
  if (i == j) return 0;
  if (i < 3 && j < 3) return 1;
  int core, ref;
  if (i < 3)      { core = i; ref = j; }
  else if (j < 3) { core = j; ref = i; }
  else return 0;
  int idx = ref - 3;
  if (idx < 3 * KK) return (idx / KK == core) ? 2 : 0;
  if (stage == 3 && idx < 6 * KK) return ((idx - 3 * KK) / KK == core) ? 3 : 0;
  return 0;
}

// ================= f32 -> bf16 weight conversion (once per launch) =================
__global__ __launch_bounds__(256)
void f32_to_bf16_kernel(const float* __restrict__ in, bf16_t* __restrict__ out, int n4) {
  int i = blockIdx.x * blockDim.x + threadIdx.x;
  if (i < n4) {
    float4 v = ((const float4*)in)[i];
    u32x2 p = {pack2(v.x, v.y), pack2(v.z, v.w)};
    *(u32x2*)(out + (size_t)i * 4) = p;
  }
}

// ================= projection GEMM: out_rows = A[M,F] @ W[F,256] + bias =================
// 64-row block per WG, 256 threads (8 waves). A staged f32->bf16 through LDS (vectorized),
// W read as bf16 B-fragments directly from global (L2-resident).
template <int F>
__global__ __launch_bounds__(256)
void proj_kernel(const float* __restrict__ A, const bf16_t* __restrict__ Wb,
                 const float* __restrict__ bias, float* __restrict__ out,
                 int M, int rpb, int nstride, int nbase)
{
  __shared__ bf16_t sA[64 * 72];    // [64][64] + pad
  const int tid  = threadIdx.x;
  const int lane = tid & 31;
  const int wv   = tid >> 5;
  const int mo   = blockIdx.x * 64;

  v8f acc[8] = {};

  for (int ko = 0; ko < F; ko += 64) {
    __syncthreads();
    // stage A chunk: 64 rows x 64 cols, float4-vectorized (4 iters/thread)
    for (int e4 = tid; e4 < 64 * 16; e4 += 256) {
      int row = e4 >> 4, c4 = (e4 & 15) << 2;
      float4 v = *(const float4*)(A + (size_t)(mo + row) * F + ko + c4);
      u32x2 p = {pack2(v.x, v.y), pack2(v.z, v.w)};
      *(u32x2*)(sA + row * 72 + c4) = p;
    }
    if (ko + 64 < F) {  // prefetch next weight rows (global_prefetch_b8)
      __builtin_prefetch(Wb + (size_t)(ko + 64) * 256 + (size_t)tid * 32, 0, 1);
    }
    __syncthreads();

    for (int ks = 0; ks < 64; ks += 32) {
      v16bf afr[4];
      for (int tm = 0; tm < 4; tm++) afr[tm] = load_a_frag(sA, 72, tm * 16, ks, lane);
      for (int jj = 0; jj < 2; jj++) {
        int tn = wv + 8 * jj;
        v16bf bfr = load_b_frag_g(Wb + (size_t)(ko + ks) * 256 + tn * 16, lane);
        for (int tm = 0; tm < 4; tm++)
          acc[jj * 4 + tm] = wmma_bf16(afr[tm], bfr, acc[jj * 4 + tm]);
      }
    }
  }

  const int half = lane >> 4;
  for (int jj = 0; jj < 2; jj++) {
    int coln = (wv + 8 * jj) * 16 + (lane & 15);
    float bv = bias[coln];
    for (int tm = 0; tm < 4; tm++)
      for (int q = 0; q < 8; q++) {
        int grow = mo + tm * 16 + q + 8 * half;     // M is a multiple of 64
        int bidx = grow / rpb, rr = grow - bidx * rpb;
        out[((size_t)bidx * nstride + nbase + rr) * 256 + coln] = acc[jj * 4 + tm][q] + bv;
      }
  }
}

// ================= fused per-batch RGAT layer =================
// One workgroup (512 threads / 16 waves) per batch element. h / Wh / alpha live in LDS
// (CDNA5 320KB/WGP); per-relation weights come in as bf16 global B-fragments.
template <int NP, int NN, int R, int STAGE>
__global__ __launch_bounds__(512)
void rgat_kernel(const float* __restrict__ nodes,
                 const bf16_t* __restrict__ GW,   // bf16 [R,256,256]
                 const float* __restrict__ Gas, const float* __restrict__ Gad,
                 float* __restrict__ out, int out_stride, int out_base)
{
  constexpr int HP = 264;           // h / Wh pitch (bf16)
  constexpr int AP = NP + 8;        // alpha pitch (bf16)
  constexpr int NT = NP / 16;       // row-tiles per wave

  extern __shared__ char smem[];
  bf16_t* sh  = (bf16_t*)smem;            // [NP][HP]
  bf16_t* sWh = sh + NP * HP;             // [NP][HP]
  bf16_t* sAl = sWh + NP * HP;            // [NP][AP]
  float*  sSrc = (float*)(sAl + NP * AP);
  float*  sDst = sSrc + NP;
  float*  sVec = sDst + NP;               // [2][256] a_src | a_dst

  const int tid    = threadIdx.x;
  const int lane   = tid & 31;
  const int wv     = tid >> 5;            // wave owns column tile tn = wv
  const int b      = blockIdx.x;
  const int half   = lane >> 4;
  const int coln16 = lane & 15;

  // ---- load h (f32 -> bf16, float4-vectorized), zero-pad rows NN..NP-1 ----
  {
    const float* hin = nodes + (size_t)b * NN * 256;
    for (int e4 = tid; e4 < NN * 64; e4 += 512) {
      float4 v = ((const float4*)hin)[e4];
      int row = e4 >> 6, c4 = (e4 & 63) << 2;
      u32x2 p = {pack2(v.x, v.y), pack2(v.z, v.w)};
      *(u32x2*)(sh + row * HP + c4) = p;
    }
    u32x2 z = {0u, 0u};
    for (int e4 = tid; e4 < (NP - NN) * 64; e4 += 512) {
      int row = NN + (e4 >> 6), c4 = (e4 & 63) << 2;
      *(u32x2*)(sh + row * HP + c4) = z;
    }
  }
  __syncthreads();

  v8f acc[NT] = {};   // output accumulators persist across relations

  for (int rel = 0; rel < R; rel++) {
    if (tid < 256) sVec[tid] = Gas[rel * 256 + tid];
    else           sVec[256 + (tid - 256)] = Gad[rel * 256 + (tid - 256)];
    const bf16_t* Wr = GW + (size_t)rel * 256 * 256;

    // ---- Wh = h @ W_rel : A-fragments from LDS, B-fragments from global bf16 ----
    v8f acc2[NT] = {};
    for (int ko = 0; ko < 256; ko += 32) {
      v16bf bfr = load_b_frag_g(Wr + (size_t)ko * 256 + wv * 16, lane);
      for (int j = 0; j < NT; j++) {
        v16bf afr = load_a_frag(sh, HP, j * 16, ko, lane);
        acc2[j] = wmma_bf16(afr, bfr, acc2[j]);
      }
    }
    __syncthreads();   // prev relation's B3 readers of sWh/sAl are done
    for (int j = 0; j < NT; j++)
      for (int q = 0; q < 8; q++)
        sWh[(j * 16 + q + 8 * half) * HP + wv * 16 + coln16] = f2bf(acc2[j][q]);
    __syncthreads();

    // ---- attention scores: s_src[n] = Wh[n,:].a_src ; s_dst[n] = Wh[n,:].a_dst ----
    if (tid < NP) {
      const unsigned* row = (const unsigned*)(sWh + tid * HP);  // 2 bf16 per dword
      float a = 0.f, d = 0.f;
      for (int e = 0; e < 128; e++) {
        unsigned u = row[e];
        float lo = __builtin_bit_cast(float, u << 16);
        float hi = __builtin_bit_cast(float, u & 0xFFFF0000u);
        a += lo * sVec[2 * e]       + hi * sVec[2 * e + 1];
        d += lo * sVec[256 + 2 * e] + hi * sVec[257 + 2 * e];
      }
      sSrc[tid] = a; sDst[tid] = d;
    }
    __syncthreads();

    // ---- masked softmax row n -> alpha (bf16, packed dword writes) ----
    if (tid < NP) {
      int n = tid, cnt = 0;
      float mx = -3.0e38f;
      if (n < NN)
        for (int m = 0; m < NN; m++)
          if (adj_label(n, m, STAGE) == rel + 1) {
            float ev = sSrc[n] + sDst[m];
            ev = ev > 0.f ? ev : 0.2f * ev;        // leaky_relu(0.2)
            mx = ev > mx ? ev : mx; cnt++;
          }
      float sum = 0.f;
      if (cnt)
        for (int m = 0; m < NN; m++)
          if (adj_label(n, m, STAGE) == rel + 1) {
            float ev = sSrc[n] + sDst[m];
            ev = ev > 0.f ? ev : 0.2f * ev;
            sum += __expf(ev - mx);
          }
      float inv = cnt ? 1.f / sum : 0.f;
      for (int m = 0; m < NP; m += 2) {
        float a0 = 0.f, a1 = 0.f;
        if (cnt && m < NN && adj_label(n, m, STAGE) == rel + 1) {
          float ev = sSrc[n] + sDst[m];
          ev = ev > 0.f ? ev : 0.2f * ev;
          a0 = __expf(ev - mx) * inv;
        }
        if (cnt && m + 1 < NN && adj_label(n, m + 1, STAGE) == rel + 1) {
          float ev = sSrc[n] + sDst[m + 1];
          ev = ev > 0.f ? ev : 0.2f * ev;
          a1 = __expf(ev - mx) * inv;
        }
        *(unsigned*)(sAl + n * AP + m) = pack2(a0, a1);
      }
    }
    __syncthreads();

    // ---- acc += alpha @ Wh ----
    for (int ko = 0; ko < NP; ko += 32) {
      v16bf bfr = load_b_frag(sWh, HP, ko, wv * 16, lane);
      for (int j = 0; j < NT; j++) {
        v16bf afr = load_a_frag(sAl, AP, j * 16, ko, lane);
        acc[j] = wmma_bf16(afr, bfr, acc[j]);
      }
    }
    // next relation's post-GEMM barrier protects sWh / sAl reuse
  }

  // ---- elu + store ----
  float* outp = out + ((size_t)b * out_stride + out_base) * 256;
  for (int j = 0; j < NT; j++) {
    int coln = wv * 16 + coln16;
    for (int q = 0; q < 8; q++) {
      int rown = j * 16 + q + 8 * half;
      if (rown < NN) {
        float v = acc[j][q];
        v = v > 0.f ? v : (__expf(v) - 1.f);
        outp[rown * 256 + coln] = v;
      }
    }
  }
}

static inline size_t rgat_lds_bytes(int NP) {
  size_t HP = 264, AP = (size_t)NP + 8;
  return (2ull * NP * HP + (size_t)NP * AP) * 2 + (2ull * NP + 512) * 4;
}

extern "C" void kernel_launch(void* const* d_in, const int* in_sizes, int n_in,
                              void* d_out, int out_size, void* d_ws, size_t ws_size,
                              hipStream_t stream) {
  const float* scene_tv   = (const float*)d_in[0];
  const float* face_tv    = (const float*)d_in[1];
  const float* text_tv    = (const float*)d_in[2];
  const float* scene_refs = (const float*)d_in[3];
  const float* face_refs  = (const float*)d_in[4];
  const float* text_refs  = (const float*)d_in[5];
  const float* scene_ra   = (const float*)d_in[6];
  const float* face_ra    = (const float*)d_in[7];
  const float* text_ra    = (const float*)d_in[8];
  const float* Ws1 = (const float*)d_in[9];  const float* bs1 = (const float*)d_in[10];
  const float* Wf1 = (const float*)d_in[11]; const float* bf1 = (const float*)d_in[12];
  const float* Wt1 = (const float*)d_in[13]; const float* bt1 = (const float*)d_in[14];
  const float* G1W = (const float*)d_in[15]; const float* G1as = (const float*)d_in[16]; const float* G1ad = (const float*)d_in[17];
  const float* Ws2 = (const float*)d_in[18]; const float* bs2 = (const float*)d_in[19];
  const float* Wf2 = (const float*)d_in[20]; const float* bf2 = (const float*)d_in[21];
  const float* Wt2 = (const float*)d_in[22]; const float* bt2 = (const float*)d_in[23];
  const float* G2W = (const float*)d_in[24]; const float* G2as = (const float*)d_in[25]; const float* G2ad = (const float*)d_in[26];
  const float* Ws3 = (const float*)d_in[27]; const float* bs3 = (const float*)d_in[28];
  const float* Wf3 = (const float*)d_in[29]; const float* bf3 = (const float*)d_in[30];
  const float* Wt3 = (const float*)d_in[31]; const float* bt3 = (const float*)d_in[32];
  const float* G3W = (const float*)d_in[33]; const float* G3as = (const float*)d_in[34]; const float* G3ad = (const float*)d_in[35];

  float* nodes1 = (float*)d_ws;                       // [B,  3, 256]
  float* nodes2 = nodes1 + (size_t)BB * 3 * 256;      // [B, 63, 256]
  float* nodes3 = nodes2 + (size_t)BB * 63 * 256;     // [B,123, 256]
  float* outF   = (float*)d_out;                      // [B,123, 256]

  // bf16 weight cache in workspace (converted every launch; deterministic)
  bf16_t* wbf = (bf16_t*)(nodes3 + (size_t)BB * 123 * 256);
  size_t woff = 0;
  auto cv = [&](const float* src, size_t n) -> bf16_t* {
    bf16_t* dst = wbf + woff; woff += n;
    int n4 = (int)(n / 4);
    f32_to_bf16_kernel<<<(n4 + 255) / 256, 256, 0, stream>>>(src, dst, n4);
    return dst;
  };
  bf16_t* Ws1b = cv(Ws1, 1024 * 256); bf16_t* Wf1b = cv(Wf1, 1024 * 256);
  bf16_t* Wt1b = cv(Wt1, 2048 * 256);
  bf16_t* Ws2b = cv(Ws2, 1024 * 256); bf16_t* Wf2b = cv(Wf2, 1024 * 256);
  bf16_t* Wt2b = cv(Wt2, 2048 * 256);
  bf16_t* Ws3b = cv(Ws3, 1024 * 256); bf16_t* Wf3b = cv(Wf3, 1024 * 256);
  bf16_t* Wt3b = cv(Wt3, 1024 * 256);
  bf16_t* G1Wb = cv(G1W, 1 * 256 * 256);
  bf16_t* G2Wb = cv(G2W, 2 * 256 * 256);
  bf16_t* G3Wb = cv(G3W, 3 * 256 * 256);

  const int Mref = BB * KK;  // 10240

  // ---- stage 1: projections -> nodes1, rgat -> nodes2 rows 0..2 ----
  proj_kernel<1024><<<BB / 64, 256, 0, stream>>>(scene_tv, Ws1b, bs1, nodes1, BB, 1, 3, 0);
  proj_kernel<1024><<<BB / 64, 256, 0, stream>>>(face_tv,  Wf1b, bf1, nodes1, BB, 1, 3, 1);
  proj_kernel<2048><<<BB / 64, 256, 0, stream>>>(text_tv,  Wt1b, bt1, nodes1, BB, 1, 3, 2);
  rgat_kernel<32, 3, 1, 1><<<BB, 512, rgat_lds_bytes(32), stream>>>(
      nodes1, G1Wb, G1as, G1ad, nodes2, 63, 0);

  // ---- stage 2: ref projections -> nodes2 rows 3..62, rgat -> nodes3 rows 0..62 ----
  proj_kernel<1024><<<Mref / 64, 256, 0, stream>>>(scene_refs, Ws2b, bs2, nodes2, Mref, KK, 63, 3);
  proj_kernel<1024><<<Mref / 64, 256, 0, stream>>>(face_refs,  Wf2b, bf2, nodes2, Mref, KK, 63, 3 + KK);
  proj_kernel<2048><<<Mref / 64, 256, 0, stream>>>(text_refs,  Wt2b, bt2, nodes2, Mref, KK, 63, 3 + 2 * KK);
  rgat_kernel<64, 63, 2, 2><<<BB, 512, rgat_lds_bytes(64), stream>>>(
      nodes2, G2Wb, G2as, G2ad, nodes3, 123, 0);

  // ---- stage 3: ra projections -> nodes3 rows 63..122, rgat -> d_out ----
  proj_kernel<1024><<<Mref / 64, 256, 0, stream>>>(scene_ra, Ws3b, bs3, nodes3, Mref, KK, 123, 63);
  proj_kernel<1024><<<Mref / 64, 256, 0, stream>>>(face_ra,  Wf3b, bf3, nodes3, Mref, KK, 123, 63 + KK);
  proj_kernel<1024><<<Mref / 64, 256, 0, stream>>>(text_ra,  Wt3b, bt3, nodes3, Mref, KK, 123, 63 + 2 * KK);
  rgat_kernel<128, 123, 3, 3><<<BB, 512, rgat_lds_bytes(128), stream>>>(
      nodes3, G3Wb, G3as, G3ad, outF, 123, 0);
}
